// GRU_84353157693987
// MI455X (gfx1250) — matrix operand order repaired and checked by reading, not compiled, not run on hardware
//
#include <hip/hip_runtime.h>

typedef __attribute__((ext_vector_type(16))) __bf16 v16bf;
typedef __attribute__((ext_vector_type(8)))  __bf16 v8bf;
typedef __attribute__((ext_vector_type(8)))  float  v8f;
typedef __attribute__((ext_vector_type(4)))  float  v4f;

constexpr int Bsz = 64;     // batch
constexpr int Ssz = 2048;   // sequence
constexpr int Isz = 256;    // input dim
constexpr int Hsz = 256;    // hidden dim
constexpr int NWG = 4;      // workgroups cooperating in the scan

__device__ __forceinline__ float sigmoid_f(float x) {
    return 1.0f / (1.0f + __expf(-x));
}
__device__ __forceinline__ float tanh_f(float x) {
    float e = __expf(2.0f * x);
    return (e - 1.0f) / (e + 1.0f);
}

// ---------------------------------------------------------------------------
// Phase 1: x_g = input @ W_g + b_g for g in {z,r,h}; output bf16, time-major
// x[g][s][b][h].  M = flattened (b,s) = 131072 rows. bf16 WMMA, f32 accum.
// ---------------------------------------------------------------------------
__global__ void __launch_bounds__(256)
gru_xgemm(const float* __restrict__ X,
          const float* __restrict__ Wz, const float* __restrict__ bz,
          const float* __restrict__ Wr, const float* __restrict__ br,
          const float* __restrict__ Wh, const float* __restrict__ bh,
          __bf16* __restrict__ xws)
{
    __shared__ __bf16 Bsw[16 * 8 * 32 * 16];   // [nt][kf][lane][16] = 128 KB

    const int g = blockIdx.y;
    const float* W    = (g == 0) ? Wz : ((g == 1) ? Wr : Wh);
    const float* bias = (g == 0) ? bz : ((g == 1) ? br : bh);

    const int tid     = threadIdx.x;
    const int wave    = tid >> 5;
    const int lane    = tid & 31;
    const int halfsel = lane >> 4;    // 0: lanes 0-15, 1: lanes 16-31
    const int lrow    = lane & 15;

    // Stage W as swizzled bf16 B-fragments: b[i] = W[kf*32 + kb + i + (i>=8 ? 8 : 0)][n]
    for (int frag = wave; frag < 128; frag += 8) {
        const int nt = frag >> 3, kf = frag & 7;
        const int kb = halfsel * 8;
        const int n  = nt * 16 + lrow;
        __bf16* dst = &Bsw[((nt * 8 + kf) * 32 + lane) * 16];
        #pragma unroll
        for (int i = 0; i < 16; ++i)
            dst[i] = (__bf16)W[(kf * 32 + kb + i + (i >= 8 ? 8 : 0)) * Hsz + n];
    }
    __syncthreads();

    // This wave's 16-row M tile (rows are consecutive s for one b: 2048%16==0)
    const int mt   = blockIdx.x * 8 + wave;
    const int m0   = mt * 16;
    const int b    = m0 / Ssz;
    const int s0   = m0 % Ssz;
    const int srow = s0 + lrow;
    const int kbA  = halfsel * 8;

    // A-fragments: lane holds K runs [kb..kb+7] and [kb+16..kb+23]
    v16bf aF[8];
    #pragma unroll
    for (int kf = 0; kf < 8; ++kf) {
        const float* src = X + (size_t)(b * Ssz + srow) * Isz + kf * 32 + kbA;
        v4f f0 = *(const v4f*)(src);
        v4f f1 = *(const v4f*)(src + 4);
        v4f f2 = *(const v4f*)(src + 16);
        v4f f3 = *(const v4f*)(src + 20);
        v16bf a;
        #pragma unroll
        for (int i = 0; i < 4; ++i) {
            a[i]      = (__bf16)f0[i];
            a[4 + i]  = (__bf16)f1[i];
            a[8 + i]  = (__bf16)f2[i];
            a[12 + i] = (__bf16)f3[i];
        }
        aF[kf] = a;
    }

    #pragma unroll 1
    for (int nt = 0; nt < 16; ++nt) {
        v8f acc = {};
        #pragma unroll
        for (int kf = 0; kf < 8; ++kf) {
            v16bf bF = *(const v16bf*)&Bsw[((nt * 8 + kf) * 32 + lane) * 16];
            acc = __builtin_amdgcn_wmma_f32_16x16x32_bf16(
                false, aF[kf], false, bF, (short)0, acc, false, false);
        }
        const int n    = nt * 16 + lrow;
        const float bv = bias[n];
        // C layout: c[r] -> M = r + halfsel*8, N = lrow
        #pragma unroll
        for (int r = 0; r < 8; ++r) {
            const int s = s0 + r + halfsel * 8;
            xws[(((size_t)g * Ssz + s) * Bsz + b) * Hsz + n] = (__bf16)(acc[r] + bv);
        }
    }
}

// ---------------------------------------------------------------------------
// Phase 2: sequential scan. 4 cooperating WGs, each owns 64 H-columns.
// U columns live pre-swizzled in LDS (96 KB); full h staged in LDS (32 KB).
// Per step: x loads issued first, 3 gate GEMMs via WMMA under their shadow,
// GRU elementwise, h chunk exchanged via double-buffered global buffer +
// device-scope atomic barrier, then async-to-LDS refresh of full h.
// ---------------------------------------------------------------------------
__global__ void __launch_bounds__(256)
gru_scan(const float* __restrict__ Uz, const float* __restrict__ buz,
         const float* __restrict__ Ur, const float* __restrict__ bur,
         const float* __restrict__ Uh, const float* __restrict__ buh,
         const __bf16* __restrict__ xws,
         float* __restrict__ out,
         __bf16* __restrict__ hbuf,          // [2][Bsz][Hsz]
         unsigned int* __restrict__ counter)
{
    __shared__ __bf16 Usw[3 * 4 * 8 * 32 * 16];  // [g][ntl][kf][lane][16] = 96 KB
    __shared__ __bf16 hA[Bsz * Hsz];             // full h, row-major, 32 KB

    const int c    = blockIdx.x;     // column chunk 0..3
    const int nb   = c * 64;
    const int tid  = threadIdx.x;
    const int wave = tid >> 5;
    const int lane = tid & 31;
    const int halfsel = lane >> 4;
    const int lrow    = lane & 15;
    const int kbA     = halfsel * 8;

    // Stage this WG's U columns as swizzled B-fragments
    for (int frag = wave; frag < 96; frag += 8) {
        const int g   = frag >> 5;             // 32 fragments per gate
        const int rem = frag & 31;
        const int ntl = rem >> 3, kf = rem & 7;
        const int kb  = halfsel * 8;
        const int n   = nb + ntl * 16 + lrow;
        const float* U = (g == 0) ? Uz : ((g == 1) ? Ur : Uh);
        __bf16* dst = &Usw[(((g * 4 + ntl) * 8 + kf) * 32 + lane) * 16];
        #pragma unroll
        for (int i = 0; i < 16; ++i)
            dst[i] = (__bf16)U[(kf * 32 + kb + i + (i >= 8 ? 8 : 0)) * Hsz + n];
    }
    // h0 = 0
    {
        v8bf z = {};
        for (int i = tid; i < (Bsz * Hsz) / 8; i += 256)
            ((v8bf*)hA)[i] = z;
    }
    __syncthreads();

    // Hoist per-tile recurrent biases (constant over t)
    float bzv[2], brv[2], bhv[2];
    int   ncol[2];
    #pragma unroll
    for (int p = 0; p < 2; ++p) {
        const int tile = wave * 2 + p;
        const int ntl  = tile & 3;
        const int n    = nb + ntl * 16 + lrow;
        ncol[p] = n;
        bzv[p] = buz[n]; brv[p] = bur[n]; bhv[p] = buh[n];
    }

    #pragma unroll 1
    for (int t = 0; t < Ssz; ++t) {
        #pragma unroll
        for (int p = 0; p < 2; ++p) {
            const int tile = wave * 2 + p;
            const int mt = tile >> 2, ntl = tile & 3;
            const int n  = ncol[p];

            if (t + 1 < Ssz) {   // prefetch next step's x for this tile
                const int mp = mt * 16 + lrow;
                __builtin_prefetch(&xws[((size_t)(0 * Ssz + t + 1) * Bsz + mp) * Hsz + n], 0, 0);
                __builtin_prefetch(&xws[((size_t)(1 * Ssz + t + 1) * Bsz + mp) * Hsz + n], 0, 0);
                __builtin_prefetch(&xws[((size_t)(2 * Ssz + t + 1) * Bsz + mp) * Hsz + n], 0, 0);
            }

            // Issue all elementwise inputs BEFORE the matmul so their latency
            // hides under the 24 WMMAs.
            float xzv[8], xrv[8], xhv[8], hpv[8];
            #pragma unroll
            for (int r = 0; r < 8; ++r) {
                const int m = mt * 16 + r + halfsel * 8;   // batch row
                xzv[r] = (float)xws[((size_t)(0 * Ssz + t) * Bsz + m) * Hsz + n];
                xrv[r] = (float)xws[((size_t)(1 * Ssz + t) * Bsz + m) * Hsz + n];
                xhv[r] = (float)xws[((size_t)(2 * Ssz + t) * Bsz + m) * Hsz + n];
                hpv[r] = (float)hA[m * Hsz + n];
            }

            v8f az = {}, ar = {}, ah = {};
            #pragma unroll
            for (int kf = 0; kf < 8; ++kf) {
                union { v16bf v; v8bf h[2]; } au;
                const int ao = (mt * 16 + lrow) * Hsz + kf * 32 + kbA;
                au.h[0] = *(const v8bf*)&hA[ao];
                au.h[1] = *(const v8bf*)&hA[ao + 16];
                const int bbase = ((ntl * 8 + kf) * 32 + lane) * 16;
                v16bf b0 = *(const v16bf*)&Usw[bbase];                      // Uz
                v16bf b1 = *(const v16bf*)&Usw[bbase + 4 * 8 * 32 * 16];    // Ur
                v16bf b2 = *(const v16bf*)&Usw[bbase + 2 * 4 * 8 * 32 * 16];// Uh
                az = __builtin_amdgcn_wmma_f32_16x16x32_bf16(false, au.v, false, b0, (short)0, az, false, false);
                ar = __builtin_amdgcn_wmma_f32_16x16x32_bf16(false, au.v, false, b1, (short)0, ar, false, false);
                ah = __builtin_amdgcn_wmma_f32_16x16x32_bf16(false, au.v, false, b2, (short)0, ah, false, false);
            }

            #pragma unroll
            for (int r = 0; r < 8; ++r) {
                const int m = mt * 16 + r + halfsel * 8;   // batch row
                const float z  = sigmoid_f(xzv[r] + az[r] + bzv[p]);
                const float rr = sigmoid_f(xrv[r] + ar[r] + brv[p]);
                const float cd = tanh_f(xhv[r] + rr * (ah[r] + bhv[p]));
                const float hn = (1.0f - z) * hpv[r] + z * cd;
                out[((size_t)m * Ssz + t) * Hsz + n] = hn;
                hbuf[(size_t)(t & 1) * (Bsz * Hsz) + m * Hsz + n] = (__bf16)hn;
            }
        }

        // Device-scope barrier across the 4 WGs (monotonic counter)
        __threadfence();
        __syncthreads();
        if (tid == 0) {
            __hip_atomic_fetch_add(counter, 1u, __ATOMIC_ACQ_REL, __HIP_MEMORY_SCOPE_AGENT);
            while (__hip_atomic_load(counter, __ATOMIC_ACQUIRE, __HIP_MEMORY_SCOPE_AGENT)
                   < (unsigned)NWG * (unsigned)(t + 1))
                __builtin_amdgcn_s_sleep(2);
        }
        __syncthreads();

        // Refresh full h_t into LDS via async-to-LDS DMA path (ASYNCcnt).
        // Each lane copies 8 x 16B => 32 KB per WG total.
        {
            const __bf16* hb = hbuf + (size_t)(t & 1) * (Bsz * Hsz);
            #pragma unroll
            for (int j = 0; j < 8; ++j) {
                unsigned ldsdst = (unsigned)(size_t)(&hA[tid * 64 + j * 8]);
                const __bf16* src = hb + tid * 64 + j * 8;
                asm volatile("global_load_async_to_lds_b128 %0, %1, off"
                             :: "v"(ldsdst), "v"(src) : "memory");
            }
            asm volatile("s_wait_asynccnt 0x0" ::: "memory");
        }
        __syncthreads();
    }
}

__global__ void gru_init(unsigned int* counter)
{
    __hip_atomic_store(counter, 0u, __ATOMIC_RELAXED, __HIP_MEMORY_SCOPE_AGENT);
}

extern "C" void kernel_launch(void* const* d_in, const int* in_sizes, int n_in,
                              void* d_out, int out_size, void* d_ws, size_t ws_size,
                              hipStream_t stream)
{
    const float* X   = (const float*)d_in[0];
    const float* Wz  = (const float*)d_in[1];
    const float* bz  = (const float*)d_in[2];
    const float* Wr  = (const float*)d_in[3];
    const float* br  = (const float*)d_in[4];
    const float* Wh  = (const float*)d_in[5];
    const float* bh  = (const float*)d_in[6];
    const float* Uz  = (const float*)d_in[7];
    const float* buz = (const float*)d_in[8];
    const float* Ur  = (const float*)d_in[9];
    const float* bur = (const float*)d_in[10];
    const float* Uh  = (const float*)d_in[11];
    const float* buh = (const float*)d_in[12];
    float* out = (float*)d_out;

    unsigned char* ws = (unsigned char*)d_ws;
    unsigned int* counter = (unsigned int*)ws;                 // 256 B
    __bf16* hbuf = (__bf16*)(ws + 256);                        // 2*64*256*2 = 64 KB
    __bf16* xws  = (__bf16*)(ws + 256 + 2 * Bsz * Hsz * 2);    // 3*S*B*H bf16 = 192 MB

    hipLaunchKernelGGL(gru_init, dim3(1), dim3(1), 0, stream, counter);
    hipLaunchKernelGGL(gru_xgemm, dim3((Bsz * Ssz) / (16 * 8), 3), dim3(256), 0, stream,
                       X, Wz, bz, Wr, br, Wh, bh, xws);
    hipLaunchKernelGGL(gru_scan, dim3(NWG), dim3(256), 0, stream,
                       Uz, buz, Ur, bur, Uh, buh, xws, out, hbuf, counter);
}
